// NeuralColony_90039694393638
// MI455X (gfx1250) — matrix operand order repaired and checked by reading, not compiled, not run on hardware
//
#include <hip/hip_runtime.h>
#include <hip/hip_bf16.h>
#include <math.h>

typedef __attribute__((ext_vector_type(16))) _Float16 v16h;
typedef __attribute__((ext_vector_type(8)))  float    v8f;

#define BB    2
#define LL    1024
#define DM    1024
#define DI    2048
#define DS    16
#define DTR   64

// ---------------------------------------------------------------------------
// WMMA GEMM, 2x2 register-blocked: each wave computes a 32x32 C macro-tile
// (4 x v_wmma_f32_16x16x32_f16 per k-chunk; A/B fragments each reused twice).
// C[M,N] = A[M,K] * B[N,K]^T   (f32 memory, f16 WMMA, f32 accumulate)
// EPI==1: C = softplus(acc + bias[n])  (dt projection epilogue)
// ---------------------------------------------------------------------------
template<int EPI>
__device__ __forceinline__
void store_tile(float* __restrict__ C, int ldc, const v8f& acc,
                int mbase, int ncol, int half, const float* __restrict__ bias)
{
#pragma unroll
    for (int r = 0; r < 8; ++r) {
        const int mrow = mbase + 8 * half + r;   // C/D layout: VGPR r -> M = 8*half + r
        float v = acc[r];
        if constexpr (EPI == 1) {
            v += bias[ncol];
            v = (v > 20.0f) ? v : log1pf(__expf(v));  // softplus
        }
        C[(size_t)mrow * ldc + ncol] = v;
    }
}

template<int EPI>
__global__ __launch_bounds__(128)
void wmma_gemm(const float* __restrict__ A, int lda,
               const float* __restrict__ Bm, int ldb,
               float* __restrict__ C, int ldc,
               int M, int N, int K,
               const float* __restrict__ bias)
{
    const int nt2    = N >> 5;                 // 32-wide macro columns
    const int nmacro = (M >> 5) * nt2;
    const int wid    = blockIdx.x * 4 + (threadIdx.x >> 5);   // wave-uniform
    if (wid >= nmacro) return;                 // uniform exit: EXEC stays all-1 for WMMA

    const int mq   = wid / nt2;
    const int nq   = wid - mq * nt2;
    const int lane = threadIdx.x & 31;
    const int l15  = lane & 15;
    const int half = lane >> 4;

    const float* Arow0 = A  + (size_t)(mq * 32 + l15) * lda;   // A row m = l&15 (ISA A-layout)
    const float* Arow1 = Arow0 + (size_t)16 * lda;
    const float* Brow0 = Bm + (size_t)(nq * 32 + l15) * ldb;   // B col n = l&15 (ISA B-layout)
    const float* Brow1 = Brow0 + (size_t)16 * ldb;

    v8f acc00 = {}, acc01 = {}, acc10 = {}, acc11 = {};
    for (int k0 = 0; k0 < K; k0 += 32) {
        if (k0 + 32 < K) {   // prefetch next k-chunk (global_prefetch_b8)
            __builtin_prefetch(Arow0 + k0 + 32, 0, 3);
            __builtin_prefetch(Arow1 + k0 + 32, 0, 3);
            __builtin_prefetch(Brow0 + k0 + 32, 0, 3);
            __builtin_prefetch(Brow1 + k0 + 32, 0, 3);
        }
        v16h a0, a1, b0, b1;
#pragma unroll
        for (int i = 0; i < 8; ++i) {
            // 16-bit A 16x32 layout: VGPR i holds K = 8*half + 2*(i&3) + 16*(i>>2), +1
            const int ka = k0 + 8 * half + 2 * (i & 3) + ((i >> 2) << 4);
            const float2 fa0 = *(const float2*)(Arow0 + ka);
            const float2 fa1 = *(const float2*)(Arow1 + ka);
            a0[2 * i] = (_Float16)fa0.x;  a0[2 * i + 1] = (_Float16)fa0.y;
            a1[2 * i] = (_Float16)fa1.x;  a1[2 * i + 1] = (_Float16)fa1.y;
            // 16-bit B 32x16 layout: VGPR i holds K = 8*half + 2*(i&3) + 16*(i>>2), +1
            const int kb = k0 + 8 * half + 2 * (i & 3) + ((i >> 2) << 4);
            const float2 fb0 = *(const float2*)(Brow0 + kb);
            const float2 fb1 = *(const float2*)(Brow1 + kb);
            b0[2 * i] = (_Float16)fb0.x;  b0[2 * i + 1] = (_Float16)fb0.y;
            b1[2 * i] = (_Float16)fb1.x;  b1[2 * i + 1] = (_Float16)fb1.y;
        }
        acc00 = __builtin_amdgcn_wmma_f32_16x16x32_f16(false, a0, false, b0, (short)0, acc00, false, false);
        acc01 = __builtin_amdgcn_wmma_f32_16x16x32_f16(false, a0, false, b1, (short)0, acc01, false, false);
        acc10 = __builtin_amdgcn_wmma_f32_16x16x32_f16(false, a1, false, b0, (short)0, acc10, false, false);
        acc11 = __builtin_amdgcn_wmma_f32_16x16x32_f16(false, a1, false, b1, (short)0, acc11, false, false);
    }

    const int n0 = nq * 32 + l15;
    const int n1 = n0 + 16;
    store_tile<EPI>(C, ldc, acc00, mq * 32,      n0, half, bias);
    store_tile<EPI>(C, ldc, acc01, mq * 32,      n1, half, bias);
    store_tile<EPI>(C, ldc, acc10, mq * 32 + 16, n0, half, bias);
    store_tile<EPI>(C, ldc, acc11, mq * 32 + 16, n1, half, bias);
}

// ---------------------------------------------------------------------------
// Depthwise causal conv (width 4) + bias + SiLU:  xm -> xc
// xm[b,l,d] = xz[(b*L+l)*2*DI + d]
// ---------------------------------------------------------------------------
__global__ __launch_bounds__(256)
void conv_silu_kernel(const float* __restrict__ xz,
                      const float* __restrict__ conv_w,
                      const float* __restrict__ conv_b,
                      float* __restrict__ xc)
{
    const int idx = blockIdx.x * blockDim.x + threadIdx.x;   // over B*L*DI
    if (idx >= BB * LL * DI) return;
    const int d = idx % DI;
    const int l = (idx / DI) % LL;
    const int b = idx / (DI * LL);

    const float* base = xz + (size_t)b * LL * (2 * DI) + d;
    float s = conv_b[d];
#pragma unroll
    for (int j = 0; j < 4; ++j) {
        const int t = l - 3 + j;
        if (t >= 0) s += base[(size_t)t * (2 * DI)] * conv_w[d * 4 + j];
    }
    xc[idx] = s / (1.0f + __expf(-s));   // silu
}

// ---------------------------------------------------------------------------
// Selective scan, one thread per (b,d); 16 states in registers.
// Fused epilogue: y = (scan_y + xc*D) * silu(z)
// ---------------------------------------------------------------------------
__global__ __launch_bounds__(256)
void scan_kernel(const float* __restrict__ proj,   // (B*L, 96): [0:64)=dt_r, [64:80)=Bc, [80:96)=Cc
                 const float* __restrict__ dt,     // (B*L, DI)
                 const float* __restrict__ xc,     // (B*L, DI)
                 const float* __restrict__ xz,     // (B*L, 2*DI), z at [DI + d]
                 const float* __restrict__ A_log,  // (DI, DS)
                 const float* __restrict__ D_param,
                 float* __restrict__ y)            // (B*L, DI)
{
    const int t = blockIdx.x * blockDim.x + threadIdx.x;
    if (t >= BB * DI) return;
    const int b = t / DI;
    const int d = t % DI;

    float Ac[DS];
#pragma unroll
    for (int n = 0; n < DS; ++n) Ac[n] = -__expf(A_log[d * DS + n]);
    const float Dp = D_param[d];

    float h[DS];
#pragma unroll
    for (int n = 0; n < DS; ++n) h[n] = 0.0f;

    for (int l = 0; l < LL; ++l) {
        const size_t row = (size_t)b * LL + l;
        const float dtv = dt[row * DI + d];
        const float xcv = xc[row * DI + d];
        const float dtx = dtv * xcv;
        const float* pr = proj + row * 96;   // uniform across lanes of a wave -> broadcast loads
        float acc = 0.0f;
#pragma unroll
        for (int n = 0; n < DS; ++n) {
            const float dA = __expf(dtv * Ac[n]);
            h[n] = h[n] * dA + dtx * pr[64 + n];
            acc += h[n] * pr[80 + n];
        }
        const float z  = xz[row * (2 * DI) + DI + d];
        const float sz = z / (1.0f + __expf(-z));
        y[row * DI + d] = (acc + xcv * Dp) * sz;
    }
}

// ---------------------------------------------------------------------------
// Colony (mean over 8 neurons of scale/bias) + residual + LayerNorm + mask.
// One block (256 thr) per (b,l) row of 1024 channels; 4 channels/thread.
// ---------------------------------------------------------------------------
__global__ __launch_bounds__(256)
void final_kernel(const float* __restrict__ x,
                  const float* __restrict__ core,
                  const float* __restrict__ scales,   // (8, DM)
                  const float* __restrict__ biases,   // (8, DM)
                  const float* __restrict__ ln_w,
                  const float* __restrict__ ln_b,
                  const float* __restrict__ mask,     // (B*L)
                  float* __restrict__ out)
{
    __shared__ float sA[8], sB[8];
    const int row = blockIdx.x;                 // 0 .. B*L-1
    const float* xr = x    + (size_t)row * DM;
    const float* cr = core + (size_t)row * DM;

    float hv[4];
    float s = 0.0f, ss = 0.0f;
#pragma unroll
    for (int i = 0; i < 4; ++i) {
        const int d = threadIdx.x + i * 256;
        float sc = 0.0f, sb = 0.0f;
#pragma unroll
        for (int n = 0; n < 8; ++n) { sc += scales[n * DM + d]; sb += biases[n * DM + d]; }
        const float v = xr[d] + (cr[d] * sc + sb) * 0.125f;
        hv[i] = v;
        s += v; ss += v * v;
    }
    // wave32 reduce
    for (int o = 16; o > 0; o >>= 1) { s += __shfl_xor(s, o); ss += __shfl_xor(ss, o); }
    const int w = threadIdx.x >> 5;
    if ((threadIdx.x & 31) == 0) { sA[w] = s; sB[w] = ss; }
    __syncthreads();
    if (threadIdx.x == 0) {
        float a = 0.0f, bsum = 0.0f;
#pragma unroll
        for (int i = 0; i < 8; ++i) { a += sA[i]; bsum += sB[i]; }
        sA[0] = a; sB[0] = bsum;
    }
    __syncthreads();
    const float mean = sA[0] * (1.0f / DM);
    const float var  = sB[0] * (1.0f / DM) - mean * mean;
    const float inv  = rsqrtf(var + 1e-5f);
    const float mk   = mask[row];
#pragma unroll
    for (int i = 0; i < 4; ++i) {
        const int d = threadIdx.x + i * 256;
        out[(size_t)row * DM + d] = ((hv[i] - mean) * inv * ln_w[d] + ln_b[d]) * mk;
    }
}

// ---------------------------------------------------------------------------
extern "C" void kernel_launch(void* const* d_in, const int* in_sizes, int n_in,
                              void* d_out, int out_size, void* d_ws, size_t ws_size,
                              hipStream_t stream)
{
    const float* x          = (const float*)d_in[0];
    const float* mask       = (const float*)d_in[1];
    const float* in_proj_w  = (const float*)d_in[2];   // (4096, 1024)
    const float* conv_w     = (const float*)d_in[3];   // (2048, 1, 4)
    const float* conv_b     = (const float*)d_in[4];
    const float* x_proj_w   = (const float*)d_in[5];   // (96, 2048)
    const float* dt_proj_w  = (const float*)d_in[6];   // (2048, 64)
    const float* dt_proj_b  = (const float*)d_in[7];
    const float* A_log      = (const float*)d_in[8];   // (2048, 16)
    const float* D_param    = (const float*)d_in[9];
    const float* out_proj_w = (const float*)d_in[10];  // (1024, 2048)
    const float* nscales    = (const float*)d_in[11];  // (8, 1024)
    const float* nbiases    = (const float*)d_in[12];
    const float* ln_w       = (const float*)d_in[13];
    const float* ln_b       = (const float*)d_in[14];
    float* out = (float*)d_out;

    float* ws   = (float*)d_ws;
    float* xz   = ws;                          // (2048, 4096)  32 MB
    float* xc   = xz   + (size_t)2048 * 4096;  // (2048, 2048)  16 MB
    float* proj = xc   + (size_t)2048 * 2048;  // (2048, 96)
    float* dt   = proj + (size_t)2048 * 96;    // (2048, 2048)  16 MB
    float* y    = dt   + (size_t)2048 * 2048;  // (2048, 2048)  16 MB
    float* core = y    + (size_t)2048 * 2048;  // (2048, 1024)   8 MB

    const int M = BB * LL;  // 2048 rows

    // 1) xz = x @ in_proj_w^T   (M=2048, N=4096, K=1024); 64*128 = 8192 macro-tiles
    {
        const int macros = (M / 32) * (4096 / 32);
        wmma_gemm<0><<<macros / 4, 128, 0, stream>>>(x, DM, in_proj_w, DM,
                                                     xz, 2 * DI, M, 2 * DI, DM, nullptr);
    }
    // 2) depthwise causal conv + SiLU -> xc
    conv_silu_kernel<<<(BB * LL * DI) / 256, 256, 0, stream>>>(xz, conv_w, conv_b, xc);

    // 3) proj = xc @ x_proj_w^T   (N=96, K=2048); 64*3 = 192 macro-tiles
    {
        const int macros = (M / 32) * (96 / 32);
        wmma_gemm<0><<<macros / 4, 128, 0, stream>>>(xc, DI, x_proj_w, DI,
                                                     proj, 96, M, 96, DI, nullptr);
    }
    // 4) dt = softplus(dt_r @ dt_proj_w^T + b)   (A = proj[:, :64], lda=96; N=2048, K=64)
    {
        const int macros = (M / 32) * (DI / 32);
        wmma_gemm<1><<<macros / 4, 128, 0, stream>>>(proj, 96, dt_proj_w, DTR,
                                                     dt, DI, M, DI, DTR, dt_proj_b);
    }
    // 5) selective scan + gating epilogue -> y
    scan_kernel<<<(BB * DI) / 256, 256, 0, stream>>>(proj, dt, xc, xz, A_log, D_param, y);

    // 6) core = y @ out_proj_w^T   (N=1024, K=2048); 64*32 = 2048 macro-tiles
    {
        const int macros = (M / 32) * (DM / 32);
        wmma_gemm<0><<<macros / 4, 128, 0, stream>>>(y, DI, out_proj_w, DI,
                                                     core, DM, M, DM, DI, nullptr);
    }
    // 7) colony + residual + LayerNorm + mask
    final_kernel<<<M, 256, 0, stream>>>(x, core, nscales, nbiases, ln_w, ln_b, mask, out);

    (void)in_sizes; (void)n_in; (void)out_size; (void)ws_size;
}
